// Custom_GroupQueryAttention_48395691491520
// MI455X (gfx1250) — matrix operand order repaired and checked by reading, not compiled, not run on hardware
//
#include <hip/hip_runtime.h>
#include <hip/hip_bf16.h>

#define T_ 1024
#define DIN 2048
#define H_ 32
#define KV_ 8
#define HD_ 64
#define GROUP_ 4

typedef __attribute__((ext_vector_type(16))) __bf16 v16bf;
typedef __attribute__((ext_vector_type(8)))  float  v8f;

union Frag16 { v16bf v; uint4 q[2]; };

__device__ __forceinline__ unsigned short f2bf(float f) {
    union { float f; unsigned u; } c; c.f = f;
    unsigned u = c.u;
    u += 0x7FFFu + ((u >> 16) & 1u);   // round-to-nearest-even
    return (unsigned short)(u >> 16);
}

// ---- CDNA5 async global->LDS DMA helpers (ASYNCcnt path) -------------------
__device__ __forceinline__ unsigned lds_off(const void* p) {
    // generic LDS pointer: low 32 bits are the DS address
    return (unsigned)(uintptr_t)p;
}
__device__ __forceinline__ void async_copy32(unsigned lds, const void* g) {
    asm volatile("global_load_async_to_lds_b128 %0, %1, off\n\t"
                 "global_load_async_to_lds_b128 %0, %1, off offset:16"
                 :: "v"(lds), "v"(g) : "memory");
}
__device__ __forceinline__ void async_copy64(unsigned lds, const void* g) {
    asm volatile("global_load_async_to_lds_b128 %0, %1, off\n\t"
                 "global_load_async_to_lds_b128 %0, %1, off offset:16\n\t"
                 "global_load_async_to_lds_b128 %0, %1, off offset:32\n\t"
                 "global_load_async_to_lds_b128 %0, %1, off offset:48"
                 :: "v"(lds), "v"(g) : "memory");
}
__device__ __forceinline__ void wait_async_le4() { asm volatile("s_wait_asynccnt 0x4" ::: "memory"); }
__device__ __forceinline__ void wait_async_0()  { asm volatile("s_wait_asynccnt 0x0" ::: "memory"); }

// ---------------------------------------------------------------- convert
__global__ __launch_bounds__(256) void tobf16_kernel(const float* __restrict__ in,
                                                     unsigned short* __restrict__ out, int n) {
    int i = blockIdx.x * 256 + threadIdx.x;
    if (i < n) out[i] = f2bf(in[i]);
}

// V: [T][KV*HD] fp32  ->  Vt: [KV*HD][T] bf16 (key-contiguous for PV B-fragments)
__global__ __launch_bounds__(256) void vtrans_kernel(const float* __restrict__ vf,
                                                     unsigned short* __restrict__ vt, int n) {
    int i = blockIdx.x * 256 + threadIdx.x;
    if (i < n) {
        int t = i >> 9;          // /512
        int c = i & 511;
        vt[(size_t)c * T_ + t] = f2bf(vf[i]);
    }
}

// ------------------------------------------------- RMSNorm + RoPE (warp per token-head)
__global__ __launch_bounds__(256) void normrope_kernel(const float* __restrict__ in,   // [T][NH*64]
                                                       unsigned short* __restrict__ out, // [NH][T][64]
                                                       const float* __restrict__ w,
                                                       const float* __restrict__ cosb,
                                                       const float* __restrict__ sinb,
                                                       int NH, float scale) {
    int warp = (blockIdx.x * 256 + threadIdx.x) >> 5;
    int lane = threadIdx.x & 31;
    int t = warp / NH, h = warp - t * NH;
    if (t >= T_) return;
    const float* xp = in + (size_t)t * NH * HD_ + h * HD_;
    float x1 = xp[lane], x2 = xp[lane + 32];
    float ss = x1 * x1 + x2 * x2;
    #pragma unroll
    for (int off = 16; off >= 1; off >>= 1) ss += __shfl_xor(ss, off, 32);
    float inv = rsqrtf(ss * (1.0f / 64.0f) + 1e-6f);
    float xn1 = x1 * inv * w[lane];
    float xn2 = x2 * inv * w[lane + 32];
    float c1 = cosb[t * HD_ + lane],      s1 = sinb[t * HD_ + lane];
    float c2 = cosb[t * HD_ + lane + 32], s2 = sinb[t * HD_ + lane + 32];
    float o1 = (xn1 * c1 - xn2 * s1) * scale;
    float o2 = (xn2 * c2 + xn1 * s2) * scale;
    unsigned short* op = out + ((size_t)h * T_ + t) * HD_;
    op[lane]      = f2bf(o1);
    op[lane + 32] = f2bf(o2);
}

// ---------------------------------------------------------------- bf16 GEMM: C = A * B^T
// A: [M][K] bf16 row-major, B: [N][K] bf16 row-major, C: [M][N] fp32.
// 128x128x32 tiles, 8 waves (4Mx2N), double-buffered async global->LDS staging.
__global__ __launch_bounds__(256) void gemm_bf16_kernel(const unsigned short* __restrict__ A,
                                                        const unsigned short* __restrict__ B,
                                                        float* __restrict__ C,
                                                        int M, int N, int K) {
    __shared__ unsigned short AsBuf[2 * 128 * 32];
    __shared__ unsigned short BsBuf[2 * 128 * 32];

    const int tid  = threadIdx.x;
    const int lane = tid & 31;
    const int ln   = lane & 15;
    const int g    = lane >> 4;
    const int wid  = tid >> 5;
    const int wm   = wid & 3;       // 0..3
    const int wn   = wid >> 2;      // 0..1
    const int m0   = blockIdx.y * 128;
    const int n0   = blockIdx.x * 128;

    v8f acc[2][4];
    #pragma unroll
    for (int mt = 0; mt < 2; ++mt)
        #pragma unroll
        for (int nt = 0; nt < 4; ++nt)
            #pragma unroll
            for (int r = 0; r < 8; ++r) acc[mt][nt][r] = 0.0f;

    const int rA = tid >> 1;          // 0..127
    const int cA = (tid & 1) * 16;    // 0 or 16
    const unsigned short* aG = A + (size_t)(m0 + rA) * K + cA;
    const unsigned short* bG = B + (size_t)(n0 + rA) * K + cA;
    unsigned ldsA[2], ldsB[2];
    ldsA[0] = lds_off(AsBuf + rA * 32 + cA);
    ldsA[1] = lds_off(AsBuf + 4096 + rA * 32 + cA);
    ldsB[0] = lds_off(BsBuf + rA * 32 + cA);
    ldsB[1] = lds_off(BsBuf + 4096 + rA * 32 + cA);

    const int NT = K >> 5;
    // prologue: stage tile 0
    async_copy32(ldsA[0], aG);
    async_copy32(ldsB[0], bG);

    for (int kt = 0; kt < NT; ++kt) {
        if (kt + 1 < NT) {
            // stage tile kt+1 into the other buffer while tile kt is consumed
            async_copy32(ldsA[(kt + 1) & 1], aG + (size_t)(kt + 1) * 32);
            async_copy32(ldsB[(kt + 1) & 1], bG + (size_t)(kt + 1) * 32);
            wait_async_le4();   // tile kt's 4 async ops complete (in-order)
        } else {
            wait_async_0();
        }
        __syncthreads();

        const unsigned short* Asb = AsBuf + (kt & 1) * 4096;
        const unsigned short* Bsb = BsBuf + (kt & 1) * 4096;

        Frag16 af[2], bf[4];
        #pragma unroll
        for (int mt = 0; mt < 2; ++mt) {
            const unsigned short* ar = Asb + (wm * 32 + mt * 16 + ln) * 32 + g * 8;
            af[mt].q[0] = *(const uint4*)(ar);
            af[mt].q[1] = *(const uint4*)(ar + 16);
        }
        #pragma unroll
        for (int nt = 0; nt < 4; ++nt) {
            const unsigned short* br = Bsb + (wn * 64 + nt * 16 + ln) * 32 + g * 8;
            bf[nt].q[0] = *(const uint4*)(br);
            bf[nt].q[1] = *(const uint4*)(br + 16);
        }
        #pragma unroll
        for (int mt = 0; mt < 2; ++mt)
            #pragma unroll
            for (int nt = 0; nt < 4; ++nt)
                acc[mt][nt] = __builtin_amdgcn_wmma_f32_16x16x32_bf16(
                    false, af[mt].v, false, bf[nt].v, (short)0, acc[mt][nt], false, false);
        __syncthreads();
    }

    #pragma unroll
    for (int mt = 0; mt < 2; ++mt)
        #pragma unroll
        for (int nt = 0; nt < 4; ++nt)
            #pragma unroll
            for (int r = 0; r < 8; ++r) {
                int m = m0 + wm * 32 + mt * 16 + r + 8 * g;
                int n = n0 + wn * 64 + nt * 16 + ln;
                C[(size_t)m * N + n] = acc[mt][nt][r];
            }
}

// ---------------------------------------------------------------- flash attention
// qb: [H][T][64] bf16 (pre-scaled by 0.125), kb: [KV][T][64] bf16, vt: [KV][64][T] bf16
// attn out: [T][H*64] bf16.  Grid: (T/128, H), 256 threads (8 waves x 16 query rows).
__global__ __launch_bounds__(256) void attn_kernel(const unsigned short* __restrict__ qb,
                                                   const unsigned short* __restrict__ kb,
                                                   const unsigned short* __restrict__ vt,
                                                   unsigned short* __restrict__ attn) {
    __shared__ unsigned short Ks[128 * 64];    // K block, row-major [key][d]
    __shared__ unsigned short Vts[64 * 128];   // V^T block, [d][key]
    __shared__ unsigned short Ps[8 * 16 * 128];// per-wave P staging

    const int tid  = threadIdx.x;
    const int lane = tid & 31;
    const int ln   = lane & 15;
    const int g    = lane >> 4;
    const int w    = tid >> 5;
    const int h    = blockIdx.y;
    const int kvh  = h >> 2;
    const int iblk = blockIdx.x;
    const int m0   = iblk * 128 + w * 16;

    // persistent Q fragments (16 rows x 64 d = 2 A-fragments)
    Frag16 qf[2];
    {
        const unsigned short* qrow = qb + ((size_t)h * T_ + (m0 + ln)) * HD_;
        #pragma unroll
        for (int kk = 0; kk < 2; ++kk) {
            int d0 = kk * 32 + g * 8;
            qf[kk].q[0] = *(const uint4*)(qrow + d0);
            qf[kk].q[1] = *(const uint4*)(qrow + d0 + 16);
        }
    }

    v8f o[4];
    #pragma unroll
    for (int nt = 0; nt < 4; ++nt)
        #pragma unroll
        for (int r = 0; r < 8; ++r) o[nt][r] = 0.0f;
    float rmax[8], rsum[8];
    #pragma unroll
    for (int r = 0; r < 8; ++r) { rmax[r] = -1e30f; rsum[r] = 0.0f; }

    // async staging addresses (per-thread): K = flat 64B chunk, Vt = [d][chunk]
    const unsigned ldsK = lds_off(Ks + tid * 32);
    const unsigned ldsV = lds_off(Vts + (tid >> 2) * 128 + (tid & 3) * 32);
    const unsigned short* kG = kb + (size_t)kvh * T_ * HD_ + tid * 32;
    const unsigned short* vG = vt + ((size_t)kvh * HD_ + (tid >> 2)) * T_ + (tid & 3) * 32;

    for (int j = 0; j <= iblk; ++j) {
        // stage K (16KB) and V^T (16KB) blocks via async DMA
        async_copy64(ldsK, kG + (size_t)j * 128 * HD_);
        async_copy64(ldsV, vG + (size_t)j * 128);
        wait_async_0();
        __syncthreads();

        // S = Q K^T  (16 x 128 per wave)
        v8f s[8];
        #pragma unroll
        for (int nt = 0; nt < 8; ++nt)
            #pragma unroll
            for (int r = 0; r < 8; ++r) s[nt][r] = 0.0f;
        #pragma unroll
        for (int kk = 0; kk < 2; ++kk)
            #pragma unroll
            for (int nt = 0; nt < 8; ++nt) {
                Frag16 bfr;
                const unsigned short* kr = Ks + (nt * 16 + ln) * 64 + kk * 32 + g * 8;
                bfr.q[0] = *(const uint4*)(kr);
                bfr.q[1] = *(const uint4*)(kr + 16);
                s[nt] = __builtin_amdgcn_wmma_f32_16x16x32_bf16(
                    false, qf[kk].v, false, bfr.v, (short)0, s[nt], false, false);
            }

        if (j == iblk) {  // causal mask on diagonal block
            #pragma unroll
            for (int nt = 0; nt < 8; ++nt) {
                int n_g = j * 128 + nt * 16 + ln;
                #pragma unroll
                for (int r = 0; r < 8; ++r) {
                    int m_g = m0 + r + 8 * g;
                    if (n_g > m_g) s[nt][r] = -1e30f;
                }
            }
        }

        // online softmax
        float nmax[8];
        #pragma unroll
        for (int r = 0; r < 8; ++r) {
            float mx = rmax[r];
            #pragma unroll
            for (int nt = 0; nt < 8; ++nt) mx = fmaxf(mx, s[nt][r]);
            #pragma unroll
            for (int off = 8; off >= 1; off >>= 1)
                mx = fmaxf(mx, __shfl_xor(mx, off, 32));
            nmax[r] = mx;
        }
        unsigned short* pw = Ps + w * (16 * 128);
        #pragma unroll
        for (int r = 0; r < 8; ++r) {
            float corr = __expf(rmax[r] - nmax[r]);
            rmax[r] = nmax[r];
            #pragma unroll
            for (int nt = 0; nt < 4; ++nt) o[nt][r] *= corr;
            float ps = 0.0f;
            #pragma unroll
            for (int nt = 0; nt < 8; ++nt) {
                float p = __expf(s[nt][r] - nmax[r]);
                s[nt][r] = p;
                ps += p;
            }
            #pragma unroll
            for (int off = 8; off >= 1; off >>= 1)
                ps += __shfl_xor(ps, off, 32);
            rsum[r] = rsum[r] * corr + ps;
        }

        // stage P (C-layout -> LDS row-major) for re-load as A fragments
        #pragma unroll
        for (int nt = 0; nt < 8; ++nt)
            #pragma unroll
            for (int r = 0; r < 8; ++r)
                pw[(r + 8 * g) * 128 + nt * 16 + ln] = f2bf(s[nt][r]);

        // O += P V   (k over 128 keys in 4 steps)
        #pragma unroll
        for (int kk = 0; kk < 4; ++kk) {
            Frag16 af;
            const unsigned short* pr = pw + ln * 128 + kk * 32 + g * 8;
            af.q[0] = *(const uint4*)(pr);
            af.q[1] = *(const uint4*)(pr + 16);
            #pragma unroll
            for (int nt = 0; nt < 4; ++nt) {
                Frag16 bfr;
                const unsigned short* vr = Vts + (nt * 16 + ln) * 128 + kk * 32 + g * 8;
                bfr.q[0] = *(const uint4*)(vr);
                bfr.q[1] = *(const uint4*)(vr + 16);
                o[nt] = __builtin_amdgcn_wmma_f32_16x16x32_bf16(
                    false, af.v, false, bfr.v, (short)0, o[nt], false, false);
            }
        }
        __syncthreads();
    }

    // epilogue: O /= rowsum, write bf16 [T][H*64]
    #pragma unroll
    for (int r = 0; r < 8; ++r) {
        float inv = 1.0f / rsum[r];
        int t = m0 + r + 8 * g;
        #pragma unroll
        for (int nt = 0; nt < 4; ++nt)
            attn[(size_t)t * (H_ * HD_) + h * HD_ + nt * 16 + ln] = f2bf(o[nt][r] * inv);
    }
}

// ---------------------------------------------------------------- launcher
extern "C" void kernel_launch(void* const* d_in, const int* in_sizes, int n_in,
                              void* d_out, int out_size, void* d_ws, size_t ws_size,
                              hipStream_t stream) {
    const float* x    = (const float*)d_in[0];
    const float* cosb = (const float*)d_in[3];
    const float* sinb = (const float*)d_in[5];
    const float* Wq   = (const float*)d_in[7];
    const float* Wk   = (const float*)d_in[8];
    const float* Wv   = (const float*)d_in[9];
    const float* Wo   = (const float*)d_in[10];
    const float* qw   = (const float*)d_in[11];
    const float* kw   = (const float*)d_in[12];
    float* out = (float*)d_out;

    char* ws = (char*)d_ws;
    size_t off = 0;
    auto alloc = [&](size_t bytes) -> char* {
        char* p = ws + off;
        off += (bytes + 255) & ~(size_t)255;
        return p;
    };
    unsigned short* xb    = (unsigned short*)alloc((size_t)T_ * DIN * 2);
    unsigned short* Wqb   = (unsigned short*)alloc((size_t)(H_ * HD_) * DIN * 2);
    unsigned short* Wkb   = (unsigned short*)alloc((size_t)(KV_ * HD_) * DIN * 2);
    unsigned short* Wvb   = (unsigned short*)alloc((size_t)(KV_ * HD_) * DIN * 2);
    unsigned short* Wob   = (unsigned short*)alloc((size_t)DIN * (H_ * HD_) * 2);
    float*          qfp   = (float*)alloc((size_t)T_ * (H_ * HD_) * 4);
    float*          kfp   = (float*)alloc((size_t)T_ * (KV_ * HD_) * 4);
    float*          vfp   = (float*)alloc((size_t)T_ * (KV_ * HD_) * 4);
    unsigned short* qb    = (unsigned short*)alloc((size_t)T_ * (H_ * HD_) * 2);
    unsigned short* kbb   = (unsigned short*)alloc((size_t)T_ * (KV_ * HD_) * 2);
    unsigned short* vtb   = (unsigned short*)alloc((size_t)T_ * (KV_ * HD_) * 2);
    unsigned short* attnb = (unsigned short*)alloc((size_t)T_ * (H_ * HD_) * 2);

    // 1) fp32 -> bf16 conversions
    {
        int n;
        n = T_ * DIN;              tobf16_kernel<<<(n + 255) / 256, 256, 0, stream>>>(x,  xb,  n);
        n = H_ * HD_ * DIN;        tobf16_kernel<<<(n + 255) / 256, 256, 0, stream>>>(Wq, Wqb, n);
        n = KV_ * HD_ * DIN;       tobf16_kernel<<<(n + 255) / 256, 256, 0, stream>>>(Wk, Wkb, n);
        n = KV_ * HD_ * DIN;       tobf16_kernel<<<(n + 255) / 256, 256, 0, stream>>>(Wv, Wvb, n);
        n = DIN * H_ * HD_;        tobf16_kernel<<<(n + 255) / 256, 256, 0, stream>>>(Wo, Wob, n);
    }

    // 2) QKV projections (WMMA GEMM, fp32 output)
    gemm_bf16_kernel<<<dim3((H_ * HD_) / 128, T_ / 128), 256, 0, stream>>>(xb, Wqb, qfp, T_, H_ * HD_, DIN);
    gemm_bf16_kernel<<<dim3((KV_ * HD_) / 128, T_ / 128), 256, 0, stream>>>(xb, Wkb, kfp, T_, KV_ * HD_, DIN);
    gemm_bf16_kernel<<<dim3((KV_ * HD_) / 128, T_ / 128), 256, 0, stream>>>(xb, Wvb, vfp, T_, KV_ * HD_, DIN);

    // 3) RMSNorm + RoPE (fold 1/sqrt(64) into Q), head-major bf16
    normrope_kernel<<<(T_ * H_  * 32) / 256, 256, 0, stream>>>(qfp, qb,  qw, cosb, sinb, H_,  0.125f);
    normrope_kernel<<<(T_ * KV_ * 32) / 256, 256, 0, stream>>>(kfp, kbb, kw, cosb, sinb, KV_, 1.0f);

    // 4) V transpose to [KV][64][T] bf16
    {
        int n = T_ * KV_ * HD_;
        vtrans_kernel<<<(n + 255) / 256, 256, 0, stream>>>(vfp, vtb, n);
    }

    // 5) flash attention
    attn_kernel<<<dim3(T_ / 128, H_), 256, 0, stream>>>(qb, kbb, vtb, attnb);

    // 6) output projection -> fp32 d_out
    gemm_bf16_kernel<<<dim3(DIN / 128, T_ / 128), 256, 0, stream>>>(attnb, Wob, out, T_, DIN, H_ * HD_);
}
